// GeneralAttention_14903536517150
// MI455X (gfx1250) — compile-verified, matrix operand order checked by reading, not run on hardware
//
#include <hip/hip_runtime.h>

// CDNA5 / gfx1250 FlashAttention-style fused attention.
// Q,K,V: fp32 [B,H,S,D]; mask: int32 [B,1,S,S]; out: fp32 [B,H,S,D]
// B=2, H=16, S=2048, D=64.  One wave (32 lanes) owns a 16-row query tile;
// 8 waves / block share LDS K/V tiles of 32 keys (f16).

typedef __attribute__((ext_vector_type(16))) _Float16 v16h;
typedef __attribute__((ext_vector_type(8)))  _Float16 v8h;
typedef __attribute__((ext_vector_type(8)))  float    v8f;

constexpr int kS  = 2048;
constexpr int kD  = 64;
constexpr int kH  = 16;
constexpr int KT  = 32;   // keys per tile (WMMA K dim for PV)
constexpr int QT  = 16;   // query rows per wave
constexpr int WV  = 8;    // waves per block
constexpr float kScale    = 0.125f;     // 1/sqrt(64)
constexpr float kMaskFill = -32768.0f;  // -(2^15)

__device__ __forceinline__ v16h pack16(const _Float16* lo, const _Float16* hi) {
    v16h r;
    #pragma unroll
    for (int j = 0; j < 8; ++j) { r[j] = lo[j]; r[j + 8] = hi[j]; }
    return r;
}

__global__ __launch_bounds__(256, 2)
void fa_fwd_kernel(const float* __restrict__ Q,
                   const float* __restrict__ K,
                   const float* __restrict__ V,
                   const int*   __restrict__ M,
                   float*       __restrict__ O)
{
    __shared__ alignas(16) _Float16 Kds[KT * kD];          // [key][d]   f16
    __shared__ alignas(16) _Float16 Vt [kD * KT];          // [d][key]   f16 (transposed)
    __shared__ alignas(16) _Float16 Pds[WV][QT * KT];      // per-wave probs

    const int tid  = threadIdx.x;
    const int wave = tid >> 5;
    const int lane = tid & 31;
    const int half = lane >> 4;   // which 16-lane half (K/M split per ISA layout)
    const int ln   = lane & 15;

    const int qblocks = kS / (WV * QT);            // 16
    const int bh      = blockIdx.x / qblocks;      // fused batch*head
    const int qblk    = blockIdx.x % qblocks;
    const int b       = bh / kH;
    const int q0      = qblk * (WV * QT) + wave * QT;

    const float* Qb = Q + (size_t)bh * kS * kD;
    const float* Kb = K + (size_t)bh * kS * kD;
    const float* Vb = V + (size_t)bh * kS * kD;
    const int*   Mb = M + (size_t)b  * kS * kS;
    float*       Ob = O + (size_t)bh * kS * kD;

    // ---- Q A-fragments: two K=32 chunks covering d = 0..63.
    // A layout (16-bit, 16x32): lane m = ln; j<8 -> k = 8*half + j,
    // j>=8 -> k = 16 + 8*half + (j-8).
    v16h qa[2];
    {
        const float* qrow = Qb + (size_t)(q0 + ln) * kD;
        #pragma unroll
        for (int c = 0; c < 2; ++c) {
            #pragma unroll
            for (int j = 0; j < 8; ++j) {
                qa[c][j]     = (_Float16)qrow[32 * c + 8 * half + j];
                qa[c][j + 8] = (_Float16)qrow[32 * c + 16 + 8 * half + j];
            }
        }
    }

    // Output accumulators: 4 d-chunks of 16; row stats for rows r + 8*half.
    v8f o[4];
    #pragma unroll
    for (int nb = 0; nb < 4; ++nb) o[nb] = v8f{};
    float m_i[8], l_i[8];
    #pragma unroll
    for (int r = 0; r < 8; ++r) { m_i[r] = -3.0e38f; l_i[r] = 0.0f; }

    // Cooperative tile-load coordinates: 256 threads x 8 contiguous elems.
    const int erow = (tid * 8) / kD;   // key_local
    const int ecol = (tid * 8) % kD;   // d start (8 contiguous)

    for (int kt0 = 0; kt0 < kS; kt0 += KT) {
        __syncthreads();               // previous-iteration LDS reads done
        {
            const float* kr = Kb + (size_t)(kt0 + erow) * kD + ecol;
            const float* vr = Vb + (size_t)(kt0 + erow) * kD + ecol;
            #pragma unroll
            for (int j = 0; j < 8; ++j) {
                Kds[erow * kD + ecol + j]  = (_Float16)kr[j];
                Vt[(ecol + j) * KT + erow] = (_Float16)vr[j];
            }
            if (kt0 + KT < kS) {       // global_prefetch_b8 for next tile
                __builtin_prefetch(kr + (size_t)KT * kD, 0, 1);
                __builtin_prefetch(vr + (size_t)KT * kD, 0, 1);
            }
        }
        __syncthreads();

        // ---- scores S = (Q K^T) for two 16-key sub-tiles (N = key)
        // B layout: lane n = key; b[j] = K[key][d(j,half,c)] -> contiguous LDS.
        v8f sc[2];
        #pragma unroll
        for (int wn = 0; wn < 2; ++wn) {
            const int key = wn * 16 + ln;
            v8f acc{};
            #pragma unroll
            for (int c = 0; c < 2; ++c) {
                v16h kb = pack16(&Kds[key * kD + 32 * c + 8 * half],
                                 &Kds[key * kD + 32 * c + 16 + 8 * half]);
                acc = __builtin_amdgcn_wmma_f32_16x16x32_f16(
                        false, qa[c], false, kb, (short)0, acc, false, false);
            }
            sc[wn] = acc;
        }

        // ---- mask + scale (C/D layout: lane holds col n = ln over rows r+8*half)
        float s0[8], s1[8];
        #pragma unroll
        for (int r = 0; r < 8; ++r) {
            const int q = q0 + r + 8 * half;
            const int* mrow = Mb + (size_t)q * kS + kt0;
            float v0 = sc[0][r] * kScale;
            float v1 = sc[1][r] * kScale;
            if (mrow[ln] == 0)      v0 = kMaskFill;
            if (mrow[16 + ln] == 0) v1 = kMaskFill;
            s0[r] = v0; s1[r] = v1;
        }

        // ---- online softmax per row; P -> per-wave LDS (f16)
        #pragma unroll
        for (int r = 0; r < 8; ++r) {
            float mx = fmaxf(s0[r], s1[r]);
            #pragma unroll
            for (int off = 8; off > 0; off >>= 1)
                mx = fmaxf(mx, __shfl_xor(mx, off, 16));
            const float mnew = fmaxf(m_i[r], mx);
            const float corr = __expf(m_i[r] - mnew);
            const float p0 = __expf(s0[r] - mnew);
            const float p1 = __expf(s1[r] - mnew);
            float rs = p0 + p1;
            #pragma unroll
            for (int off = 8; off > 0; off >>= 1)
                rs += __shfl_xor(rs, off, 16);
            l_i[r] = l_i[r] * corr + rs;
            m_i[r] = mnew;
            #pragma unroll
            for (int nb = 0; nb < 4; ++nb) o[nb][r] *= corr;
            const int prow = r + 8 * half;
            Pds[wave][prow * KT + ln]      = (_Float16)p0;
            Pds[wave][prow * KT + 16 + ln] = (_Float16)p1;
        }
        __builtin_amdgcn_wave_barrier();   // same-wave LDS ops are in-order

        // ---- O += P x V  (A = P 16x32, B = Vt column reads are contiguous)
        v16h pa = pack16(&Pds[wave][ln * KT + 8 * half],
                         &Pds[wave][ln * KT + 16 + 8 * half]);
        #pragma unroll
        for (int nb = 0; nb < 4; ++nb) {
            const int d = nb * 16 + ln;
            v16h vb = pack16(&Vt[d * KT + 8 * half],
                             &Vt[d * KT + 16 + 8 * half]);
            o[nb] = __builtin_amdgcn_wmma_f32_16x16x32_f16(
                        false, pa, false, vb, (short)0, o[nb], false, false);
        }
    }

    // ---- normalize and store
    #pragma unroll
    for (int r = 0; r < 8; ++r) {
        const float inv = 1.0f / l_i[r];
        const int q = q0 + r + 8 * half;
        #pragma unroll
        for (int nb = 0; nb < 4; ++nb) {
            const int d = nb * 16 + ln;
            Ob[(size_t)q * kD + d] = o[nb][r] * inv;
        }
    }
}

extern "C" void kernel_launch(void* const* d_in, const int* in_sizes, int n_in,
                              void* d_out, int out_size, void* d_ws, size_t ws_size,
                              hipStream_t stream) {
    (void)n_in; (void)out_size; (void)d_ws; (void)ws_size;
    const float* Q = (const float*)d_in[0];
    const float* K = (const float*)d_in[1];
    const float* V = (const float*)d_in[2];
    const int*   M = (const int*)d_in[3];
    float*       O = (float*)d_out;

    const int BH      = in_sizes[0] / (kS * kD);     // B*H = 32
    const int qblocks = kS / (WV * QT);              // 16
    dim3 grid(BH * qblocks);
    dim3 block(256);
    fa_fwd_kernel<<<grid, block, 0, stream>>>(Q, K, V, M, O);
}
